// WaveNetBackbone_26774826123410
// MI455X (gfx1250) — matrix operand order repaired and checked by reading, not compiled, not run on hardware
//
#include <hip/hip_runtime.h>

typedef __attribute__((ext_vector_type(16))) _Float16 v16h;
typedef __attribute__((ext_vector_type(8)))  _Float16 v8h;
typedef __attribute__((ext_vector_type(8)))  float    v8f;

#define L_TOT 20
#define NLAY  10
#define CCH   32
#define SCH   32
#define BB    4
#define TT    32768
#define TILES_PER_B (TT / 16)            // 2048
#define N_TILES     (BB * TILES_PER_B)   // 8192
#define NWG 512
#define WAVES_PER_WG 4
#define BLOCK (WAVES_PER_WG * 32)        // 128 threads = 4 wave32
#define TOTAL_WAVES (NWG * WAVES_PER_WG) // 2048
#define TPW (N_TILES / TOTAL_WAVES)      // 4 tiles (of 16 timesteps) per wave

__device__ __forceinline__ v8f wmma16(v16h a, v16h b, v8f c) {
  // D = A(16x32 f16) * B(32x16 f16) + C(16x16 f32)
  return __builtin_amdgcn_wmma_f32_16x16x32_f16(false, a, false, b, (short)0, c,
                                                false, false);
}

__device__ __forceinline__ v8f zero8() {
  v8f z;
#pragma unroll
  for (int j = 0; j < 8; ++j) z[j] = 0.0f;
  return z;
}

// ---------------------------------------------------------------------------
// Pack fp32 weights into the CDNA5 16-bit A-matrix lane layout (f16).
// wpack entry index = ((l*5 + mat)*2 + tile)*32 + lane, 16 halves (32B) each.
// mat: 0 = w_dil tap0, 1 = w_dil tap1, 2 = w_tanh, 3 = w_sig, 4 = w_skip
// A layout as packed here (16x32, contiguous k, B matches): lane<16 row M=lane&15 holds K={0..15},
// lane>=16 holds K={16..31}; halves j=0..15 map k = j + 16*(lane>=16).
// ---------------------------------------------------------------------------
__global__ void wavenet_pack_weights(const float* __restrict__ w_dil,
                                     const float* __restrict__ w_tanh,
                                     const float* __restrict__ w_sig,
                                     const float* __restrict__ w_skip,
                                     _Float16* __restrict__ wpack) {
  int id = blockIdx.x * blockDim.x + threadIdx.x;
  if (id >= L_TOT * 5 * 2 * 32) return;
  int lane = id & 31;
  int tile = (id >> 5) & 1;
  int mat  = (id >> 6) % 5;
  int l    = id / (5 * 2 * 32);
  int m = tile * 16 + (lane & 15);
  _Float16 vals[16];
#pragma unroll
  for (int j = 0; j < 16; ++j) {
    int k = j + ((lane >= 16) ? 16 : 0);
    float w;
    if (mat == 0)      w = w_dil[((size_t)(l * CCH + m) * CCH + k) * 2 + 0];
    else if (mat == 1) w = w_dil[((size_t)(l * CCH + m) * CCH + k) * 2 + 1];
    else if (mat == 2) w = w_tanh[(size_t)(l * CCH + m) * CCH + k];
    else if (mat == 3) w = w_sig[(size_t)(l * CCH + m) * CCH + k];
    else               w = w_skip[(size_t)(l * SCH + m) * CCH + k];
    vals[j] = (_Float16)w;
  }
  ((v16h*)wpack)[id] = *(v16h*)vals;
}

// ---------------------------------------------------------------------------
// Input stem: h0[b][t][c] = f16(w_in[c] * x[b,0,t] + b_in[c]), time-major f16.
// Also zeroes the grid-barrier counter (runs before the main kernel).
// ---------------------------------------------------------------------------
__global__ void wavenet_stem(const float* __restrict__ x,
                             const float* __restrict__ w_in,
                             const float* __restrict__ b_in,
                             _Float16* __restrict__ h0,
                             int* __restrict__ counter) {
  int id = blockIdx.x * blockDim.x + threadIdx.x; // b*T + t
  if (id == 0) *counter = 0;
  if (id >= BB * TT) return;
  float xv = x[id];
  _Float16 hv[32];
#pragma unroll
  for (int c = 0; c < CCH; ++c) hv[c] = (_Float16)(w_in[c] * xv + b_in[c]);
  v16h* dst = (v16h*)(h0 + (size_t)id * CCH);
  dst[0] = *(v16h*)(hv);
  dst[1] = *(v16h*)(hv + 16);
}

// ---------------------------------------------------------------------------
// Persistent main kernel: each wave owns TPW tiles of 16 timesteps for all
// 20 layers. skip accumulates in LDS; h ping-pongs between two global f16
// buffers; layers separated by a monotonic-counter grid barrier.
// ---------------------------------------------------------------------------
__global__ __launch_bounds__(BLOCK) void wavenet_main(
    const float* __restrict__ b_dil, const float* __restrict__ b_tanh,
    const float* __restrict__ b_sig, const float* __restrict__ b_skip,
    const _Float16* __restrict__ wpack, _Float16* __restrict__ hws0,
    _Float16* __restrict__ hws1, float* __restrict__ out,
    int* __restrict__ counter) {
  __shared__ float    skip_lds[WAVES_PER_WG * TPW * 16 * SCH]; // 32 KB
  __shared__ _Float16 stage[WAVES_PER_WG * 16 * CCH];          // 4 KB

  const int lane  = threadIdx.x & 31;
  const int wib   = threadIdx.x >> 5;
  const int gwave = blockIdx.x * WAVES_PER_WG + wib;
  const int lhalf = lane & 15;
  const int lhi   = lane >> 4;  // 0 or 1
  const int mb    = lhi * 8;    // accumulator row base within a 16-row tile

  float*    myskip  = skip_lds + wib * (TPW * 16 * SCH);
  _Float16* mystage = stage + wib * (16 * CCH);

  for (int i = lane; i < TPW * 16 * SCH; i += 32) myskip[i] = 0.0f;

  _Float16* hsrc = hws0;
  _Float16* hdst = hws1;

  for (int l = 0; l < L_TOT; ++l) {
    const int d = 1 << (l % NLAY);

    // ---- per-layer A-tiles (5 matrices x 2 M-tiles), one 32B load each ----
    v16h W[10];
    const v16h* wp = ((const v16h*)wpack) + (size_t)l * (10 * 32);
#pragma unroll
    for (int m = 0; m < 10; ++m) W[m] = wp[m * 32 + lane];

    // ---- per-lane biases for the accumulator rows this lane owns ----
    float bd0[8], bd1[8], bt0[8], bt1[8], bs0[8], bs1[8];
#pragma unroll
    for (int r = 0; r < 8; ++r) {
      bd0[r] = b_dil [l * CCH + mb + r];  bd1[r] = b_dil [l * CCH + 16 + mb + r];
      bt0[r] = b_tanh[l * CCH + mb + r];  bt1[r] = b_tanh[l * CCH + 16 + mb + r];
      bs0[r] = b_sig [l * CCH + mb + r];  bs1[r] = b_sig [l * CCH + 16 + mb + r];
    }

    for (int i = 0; i < TPW; ++i) {
      const int tile = gwave * TPW + i;
      const int b    = tile / TILES_PER_B;
      const int t0   = (tile % TILES_PER_B) * 16;
      const int t    = t0 + lhalf;
      const size_t rowbase = ((size_t)b * TT + t) * CCH;

      // B operands: current h[t] and shifted h[t-d] (zero below t=0)
      v16h Bc = *(const v16h*)(hsrc + rowbase + lhi * 16);
      v16h Bs;
#pragma unroll
      for (int j = 0; j < 16; ++j) Bs[j] = (_Float16)0.0f;
      if (t >= d)
        Bs = *(const v16h*)(hsrc + rowbase - (size_t)d * CCH + lhi * 16);

      // xd = W0 @ h[t-d] + W1 @ h[t] + b_dil    (2 M-tiles, 4 WMMAs)
      v8f a0 = zero8(), a1 = zero8();
      a0 = wmma16(W[0], Bs, a0);
      a0 = wmma16(W[2], Bc, a0);
      a1 = wmma16(W[1], Bs, a1);
      a1 = wmma16(W[3], Bc, a1);

      float xd0[8], xd1[8];
#pragma unroll
      for (int r = 0; r < 8; ++r) { xd0[r] = a0[r] + bd0[r]; xd1[r] = a1[r] + bd1[r]; }

      // restage xd (acc layout) -> LDS [t][c] f16 -> reload in B layout
      {
        v8h p0, p1;
#pragma unroll
        for (int r = 0; r < 8; ++r) { p0[r] = (_Float16)xd0[r]; p1[r] = (_Float16)xd1[r]; }
        *(v8h*)(mystage + lhalf * CCH + mb)      = p0;
        *(v8h*)(mystage + lhalf * CCH + 16 + mb) = p1;
      }
      v16h Bxd = *(const v16h*)(mystage + lhalf * CCH + lhi * 16);

      // f = tanh(Wt@xd + bt), g = sigmoid(Ws@xd + bs)   (4 WMMAs)
      v8f f0 = wmma16(W[4], Bxd, zero8());
      v8f f1 = wmma16(W[5], Bxd, zero8());
      v8f g0 = wmma16(W[6], Bxd, zero8());
      v8f g1 = wmma16(W[7], Bxd, zero8());

      float xh0[8], xh1[8];
#pragma unroll
      for (int r = 0; r < 8; ++r) {
        float e0  = __expf(2.0f * (f0[r] + bt0[r]));
        float th0 = (e0 - 1.0f) / (e0 + 1.0f);
        float sg0 = 1.0f / (1.0f + __expf(-(g0[r] + bs0[r])));
        xh0[r] = th0 * sg0;
        float e1  = __expf(2.0f * (f1[r] + bt1[r]));
        float th1 = (e1 - 1.0f) / (e1 + 1.0f);
        float sg1 = 1.0f / (1.0f + __expf(-(g1[r] + bs1[r])));
        xh1[r] = th1 * sg1;
      }

      // restage xh -> LDS -> B layout
      {
        v8h p0, p1;
#pragma unroll
        for (int r = 0; r < 8; ++r) { p0[r] = (_Float16)xh0[r]; p1[r] = (_Float16)xh1[r]; }
        *(v8h*)(mystage + lhalf * CCH + mb)      = p0;
        *(v8h*)(mystage + lhalf * CCH + 16 + mb) = p1;
      }
      v16h Bxh = *(const v16h*)(mystage + lhalf * CCH + lhi * 16);

      // skip += W_skip @ xh   (2 WMMAs), accumulate in LDS
      v8f s0 = wmma16(W[8], Bxh, zero8());
      v8f s1 = wmma16(W[9], Bxh, zero8());
      float* sk = myskip + i * (16 * SCH) + lhalf * SCH;
#pragma unroll
      for (int r = 0; r < 8; ++r) {
        sk[mb + r]      += s0[r];
        sk[16 + mb + r] += s1[r];
      }

      // residual h_next = xh + xd -> global f16 (time-major)
      {
        v8h p0, p1;
#pragma unroll
        for (int r = 0; r < 8; ++r) {
          p0[r] = (_Float16)(xh0[r] + xd0[r]);
          p1[r] = (_Float16)(xh1[r] + xd1[r]);
        }
        *(v8h*)(hdst + rowbase + mb)      = p0;
        *(v8h*)(hdst + rowbase + 16 + mb) = p1;
      }
    }

    // -------- device-wide barrier between layers (monotonic counter) -------
    __threadfence();
    __syncthreads();
    if (threadIdx.x == 0) {
      __hip_atomic_fetch_add(counter, 1, __ATOMIC_RELEASE,
                             __HIP_MEMORY_SCOPE_AGENT);
      const int target = (l + 1) * NWG;
      while (__hip_atomic_load(counter, __ATOMIC_ACQUIRE,
                               __HIP_MEMORY_SCOPE_AGENT) < target) {
        __builtin_amdgcn_s_sleep(8);
      }
    }
    __syncthreads();

    _Float16* tmp = hdst; hdst = hsrc; hsrc = tmp;
  }

  // -------- writeout: skip_sum + sum_l b_skip, coalesced over t ------------
  float bsum[16];
#pragma unroll
  for (int rr = 0; rr < 16; ++rr) {
    int s = rr * 2 + lhi;
    float acc = 0.0f;
    for (int l = 0; l < L_TOT; ++l) acc += b_skip[l * SCH + s];
    bsum[rr] = acc;
  }
  for (int i = 0; i < TPW; ++i) {
    const int tile = gwave * TPW + i;
    const int b    = tile / TILES_PER_B;
    const int t0   = (tile % TILES_PER_B) * 16;
    const float* sk = myskip + i * (16 * SCH);
#pragma unroll
    for (int rr = 0; rr < 16; ++rr) {
      int s = rr * 2 + lhi;
      out[((size_t)(b * SCH + s)) * TT + t0 + lhalf] =
          sk[lhalf * SCH + s] + bsum[rr];
    }
  }
}

extern "C" void kernel_launch(void* const* d_in, const int* in_sizes, int n_in,
                              void* d_out, int out_size, void* d_ws,
                              size_t ws_size, hipStream_t stream) {
  const float* x      = (const float*)d_in[0];
  const float* w_in   = (const float*)d_in[1];
  const float* b_in   = (const float*)d_in[2];
  const float* w_dil  = (const float*)d_in[3];
  const float* b_dil  = (const float*)d_in[4];
  const float* w_tanh = (const float*)d_in[5];
  const float* b_tanh = (const float*)d_in[6];
  const float* w_sig  = (const float*)d_in[7];
  const float* b_sig  = (const float*)d_in[8];
  const float* w_skip = (const float*)d_in[9];
  const float* b_skip = (const float*)d_in[10];
  float* out = (float*)d_out;

  char* ws = (char*)d_ws;
  int*      counter = (int*)ws;                         // 4 B (zeroed by stem)
  _Float16* wpack   = (_Float16*)(ws + 256);            // 200 KB packed weights
  _Float16* h0      = (_Float16*)(ws + (512 << 10));    // 8 MB f16 activations
  _Float16* h1      = (_Float16*)(ws + (512 << 10) + (size_t)BB * TT * CCH * 2);

  wavenet_pack_weights<<<dim3((L_TOT * 5 * 2 * 32) / 256), dim3(256), 0,
                         stream>>>(w_dil, w_tanh, w_sig, w_skip, wpack);
  wavenet_stem<<<dim3((BB * TT) / 256), dim3(256), 0, stream>>>(x, w_in, b_in,
                                                                h0, counter);
  wavenet_main<<<dim3(NWG), dim3(BLOCK), 0, stream>>>(
      b_dil, b_tanh, b_sig, b_skip, wpack, h0, h1, out, counter);
}
